// LSHmodule_41893111005398
// MI455X (gfx1250) — matrix operand order repaired and checked by reading, not compiled, not run on hardware
//
#include <hip/hip_runtime.h>
#include <hip/hip_bf16.h>
#include <math.h>

// ---------------------------------------------------------------------------
// LSH attention for MI455X (gfx1250, wave32, WMMA bf16 16x16x32, f32 accum)
//   B=4, N=2048, D=512, H=8, e=64, N_HYPER=5, scale = 1/sqrt(512)
// ---------------------------------------------------------------------------

typedef __bf16 bf16_t;
typedef __attribute__((ext_vector_type(8)))  __bf16 v8bf;
typedef __attribute__((ext_vector_type(16))) __bf16 v16bf;
typedef __attribute__((ext_vector_type(8)))  float  v8f;

#define BATCH 4
#define SEQ   2048
#define DIM   512
#define HEADS 8
#define EDIM  64
#define SCALE 0.044194173824159216f   // 1/sqrt(512)

static __device__ inline v8bf ld8(const bf16_t* p) {
    return *reinterpret_cast<const v8bf*>(p);
}

static __device__ inline v16bf combine(v8bf lo, v8bf hi) {
    v16bf r;
#pragma unroll
    for (int i = 0; i < 8; ++i) { r[i] = lo[i]; r[i + 8] = hi[i]; }
    return r;
}

static __device__ inline v8f wmma_bf16(v16bf a, v16bf b, v8f c) {
    // D(f32 16x16) = A(bf16 16x32) * B(bf16 32x16) + C
    return __builtin_amdgcn_wmma_f32_16x16x32_bf16(
        false, a, false, b, (short)0, c, false, false);
}

// ---------------------------------------------------------------------------
// f32 -> bf16 conversion
// ---------------------------------------------------------------------------
__global__ __launch_bounds__(256) void cvt_bf16_kernel(
    const float* __restrict__ src, bf16_t* __restrict__ dst, int n) {
    int i = blockIdx.x * 256 + threadIdx.x;
    if (i < n) dst[i] = (bf16_t)src[i];
}

// ---------------------------------------------------------------------------
// GEMM: out[m,d] = sum_k x[m,k] * W[d,k] + bias[d]
// Each wave computes a 16(M) x 64(D) tile; K-loop step 32 -> 4 WMMAs/step.
// Software-pipelined: next K-step fragments are loaded before the current
// step's WMMAs so waits are partial-count instead of s_wait_loadcnt 0.
// (The last-step over-read of <=128B lands in adjacent workspace regions.)
// mode 0: write q f32 (for LSH signs) + q bf16 (for attention)
// mode 1: write v bf16 transposed: vT[((b*H+h)*EDIM + eidx)*SEQ + n]
// ---------------------------------------------------------------------------
__global__ __launch_bounds__(256) void gemm_bias_kernel(
    const bf16_t* __restrict__ xb, const bf16_t* __restrict__ wb,
    const float* __restrict__ bias,
    float* __restrict__ qf, bf16_t* __restrict__ qb,
    bf16_t* __restrict__ vT, int mode) {
    const int lane = threadIdx.x & 31;
    const int wv   = threadIdx.x >> 5;
    const int hi   = lane >> 4;       // half-wave select
    const int n16  = lane & 15;

    const int id    = blockIdx.x * 8 + wv;   // 0..4095
    const int mtile = id >> 3;               // 0..511
    const int dtile = id & 7;                // 0..7
    const int m0 = mtile * 16;
    const int d0 = dtile * 64;

    v8f acc0 = {}, acc1 = {}, acc2 = {}, acc3 = {};

    const bf16_t* arow = xb + (m0 + n16) * DIM;               // A: row = lane&15
    const bf16_t* w0 = wb + (d0 +  0 + n16) * DIM;            // B: col = lane&15
    const bf16_t* w1 = wb + (d0 + 16 + n16) * DIM;
    const bf16_t* w2 = wb + (d0 + 32 + n16) * DIM;
    const bf16_t* w3 = wb + (d0 + 48 + n16) * DIM;

    // ---- prologue loads (k0 = 0) -------------------------------------------
    // A frag: lane row m=lane&15; elem i -> k = hi*16 + i (contiguous 16)
    v16bf a  = combine(ld8(arow + hi * 16), ld8(arow + hi * 16 + 8));
    // B frag: lane col n=lane&15; elem i -> k = hi*16 + i (contiguous 16)
    v16bf b0 = combine(ld8(w0 + hi * 16), ld8(w0 + hi * 16 + 8));
    v16bf b1 = combine(ld8(w1 + hi * 16), ld8(w1 + hi * 16 + 8));
    v16bf b2 = combine(ld8(w2 + hi * 16), ld8(w2 + hi * 16 + 8));
    v16bf b3 = combine(ld8(w3 + hi * 16), ld8(w3 + hi * 16 + 8));

    for (int k0 = 0; k0 < DIM; k0 += 32) {
        const int kn = k0 + 32;   // last-iter preload over-reads into workspace
        __builtin_prefetch(arow + kn + 32, 0, 3);
        v16bf an  = combine(ld8(arow + kn + hi * 16), ld8(arow + kn + hi * 16 + 8));
        v16bf bn0 = combine(ld8(w0 + kn + hi * 16), ld8(w0 + kn + hi * 16 + 8));
        v16bf bn1 = combine(ld8(w1 + kn + hi * 16), ld8(w1 + kn + hi * 16 + 8));
        v16bf bn2 = combine(ld8(w2 + kn + hi * 16), ld8(w2 + kn + hi * 16 + 8));
        v16bf bn3 = combine(ld8(w3 + kn + hi * 16), ld8(w3 + kn + hi * 16 + 8));

        acc0 = wmma_bf16(a, b0, acc0);
        acc1 = wmma_bf16(a, b1, acc1);
        acc2 = wmma_bf16(a, b2, acc2);
        acc3 = wmma_bf16(a, b3, acc3);

        a = an; b0 = bn0; b1 = bn1; b2 = bn2; b3 = bn3;
    }

    float bs0 = bias[d0 +  0 + n16];
    float bs1 = bias[d0 + 16 + n16];
    float bs2 = bias[d0 + 32 + n16];
    float bs3 = bias[d0 + 48 + n16];

    if (mode == 0) {
#pragma unroll
        for (int r = 0; r < 8; ++r) {
            int row = m0 + r + 8 * hi;        // C layout: VGPR r -> M = r + 8*hi
            float v0 = acc0[r] + bs0, v1 = acc1[r] + bs1;
            float v2 = acc2[r] + bs2, v3 = acc3[r] + bs3;
            float*  qr = qf + row * DIM + d0;
            bf16_t* br = qb + row * DIM + d0;
            qr[ 0 + n16] = v0; qr[16 + n16] = v1;
            qr[32 + n16] = v2; qr[48 + n16] = v3;
            br[ 0 + n16] = (bf16_t)v0; br[16 + n16] = (bf16_t)v1;
            br[32 + n16] = (bf16_t)v2; br[48 + n16] = (bf16_t)v3;
        }
    } else {
        const int h = dtile;                   // d0 = h*64
#pragma unroll
        for (int r = 0; r < 8; ++r) {
            int row = m0 + r + 8 * hi;
            int b   = row >> 11;               // /2048
            int nn  = row & 2047;
            bf16_t* vb = vT + ((b * HEADS + h) * EDIM) * SEQ + nn;
            vb[( 0 + n16) * SEQ] = (bf16_t)(acc0[r] + bs0);
            vb[(16 + n16) * SEQ] = (bf16_t)(acc1[r] + bs1);
            vb[(32 + n16) * SEQ] = (bf16_t)(acc2[r] + bs2);
            vb[(48 + n16) * SEQ] = (bf16_t)(acc3[r] + bs3);
        }
    }
}

// ---------------------------------------------------------------------------
// LSH buckets: proj = [q_head, 1] @ hyperplanes(65x5); bits = proj>=0
// bucket = sum 2^y * bit_y ; stored as bkt[(b*H+h)*SEQ + n]
// ---------------------------------------------------------------------------
__global__ __launch_bounds__(256) void buckets_kernel(
    const float* __restrict__ qf, const float* __restrict__ hyp,
    int* __restrict__ bkt) {
    int idx = blockIdx.x * 256 + threadIdx.x;     // B*N*H = 65536
    if (idx >= BATCH * SEQ * HEADS) return;
    int h = idx & 7;
    int n = (idx >> 3) & (SEQ - 1);
    int b = idx >> 14;
    const float* q = qf + (b * SEQ + n) * DIM + h * EDIM;
    float acc[5];
#pragma unroll
    for (int y = 0; y < 5; ++y) acc[y] = hyp[64 * 5 + y];   // bias coord (==1)
    for (int e = 0; e < EDIM; ++e) {
        float qe = q[e];
#pragma unroll
        for (int y = 0; y < 5; ++y) acc[y] += qe * hyp[e * 5 + y];
    }
    int bucket = 0;
#pragma unroll
    for (int y = 0; y < 5; ++y)
        if (acc[y] >= 0.f) bucket |= (1 << y);
    bkt[(b * HEADS + h) * SEQ + n] = bucket;
}

// ---------------------------------------------------------------------------
// Fused bucket-masked attention (flash-style streaming softmax).
// Masked scores are 0 (NOT -inf), exactly like the reference, so they simply
// participate in the streaming softmax with value 0.
// Block = 256 threads = 8 waves; each wave owns 16 rows of one (b,h).
// Column loop step = 32 -> 4 score WMMAs + 4 P@V WMMAs per step.
// Pipelined: V frags load before the softmax; next tile's Q^T frags are
// prefetched during the softmax + P@V phase.
// ---------------------------------------------------------------------------
__global__ __launch_bounds__(256) void lsh_attn_kernel(
    const bf16_t* __restrict__ qb, const bf16_t* __restrict__ vT,
    const int* __restrict__ bkt, float* __restrict__ out) {
    __shared__ __align__(16) bf16_t Pt[8][16 * 32];   // per-wave P tile (8 KB)

    const int lane = threadIdx.x & 31;
    const int wv   = threadIdx.x >> 5;
    const int hi   = lane >> 4;
    const int n16  = lane & 15;

    const int bh    = blockIdx.x >> 4;     // 0..31  (b*H + h)
    const int rblk  = blockIdx.x & 15;     // 0..15
    const int b     = bh >> 3;
    const int h     = bh & 7;
    const int rbase = rblk * 128 + wv * 16;

    // Q row A-fragments, K = 0..63 (e-dim), loaded once
    const bf16_t* qrow = qb + ((b * SEQ + rbase + n16) * DIM + h * EDIM);
    v16bf aq0 = combine(ld8(qrow +      hi * 16), ld8(qrow +      hi * 16 + 8));
    v16bf aq1 = combine(ld8(qrow + 32 + hi * 16), ld8(qrow + 32 + hi * 16 + 8));

    int rbk[8];
#pragma unroll
    for (int r = 0; r < 8; ++r)
        rbk[r] = bkt[bh * SEQ + rbase + r + 8 * hi];

    float mrow[8], lrow[8];
#pragma unroll
    for (int r = 0; r < 8; ++r) { mrow[r] = 0.f; lrow[r] = 0.f; }
    v8f O0 = {}, O1 = {}, O2 = {}, O3 = {};

    const bf16_t* vbh = vT + (bh * EDIM) * SEQ;

    // ---- prologue: Q^T B-frags for the first column tile -------------------
    const bf16_t* c0 = qb + ((b * SEQ + n16) * DIM + h * EDIM);
    const bf16_t* c1 = c0 + 16 * DIM;
    v16bf b00 = combine(ld8(c0 +      hi * 16), ld8(c0 +      hi * 16 + 8));
    v16bf b01 = combine(ld8(c0 + 32 + hi * 16), ld8(c0 + 32 + hi * 16 + 8));
    v16bf b10 = combine(ld8(c1 +      hi * 16), ld8(c1 +      hi * 16 + 8));
    v16bf b11 = combine(ld8(c1 + 32 + hi * 16), ld8(c1 + 32 + hi * 16 + 8));

    for (int j0 = 0; j0 < SEQ; j0 += 32) {
        // ---- scores: S = Q * Q^T for 16 rows x 32 cols (two 16x16 tiles) ----
        v8f S0 = {}, S1 = {};
        S0 = wmma_bf16(aq0, b00, S0);
        S0 = wmma_bf16(aq1, b01, S0);
        S1 = wmma_bf16(aq0, b10, S1);
        S1 = wmma_bf16(aq1, b11, S1);

        // ---- V B-frags: independent of softmax, issue loads now ------------
        const bf16_t* v0 = vbh + ( 0 + n16) * SEQ + j0 + hi * 16;
        const bf16_t* v1 = vbh + (16 + n16) * SEQ + j0 + hi * 16;
        const bf16_t* v2 = vbh + (32 + n16) * SEQ + j0 + hi * 16;
        const bf16_t* v3 = vbh + (48 + n16) * SEQ + j0 + hi * 16;
        v16bf vb0 = combine(ld8(v0), ld8(v0 + 8));
        v16bf vb1 = combine(ld8(v1), ld8(v1 + 8));
        v16bf vb2 = combine(ld8(v2), ld8(v2 + 8));
        v16bf vb3 = combine(ld8(v3), ld8(v3 + 8));

        // ---- prefetch next tile's Q^T B-frags (hidden under softmax) -------
        // (last-iter over-read lands in the vT workspace region: safe)
        const bf16_t* n0 = qb + ((b * SEQ + j0 + 32 + n16) * DIM + h * EDIM);
        const bf16_t* n1 = n0 + 16 * DIM;
        v16bf nb00 = combine(ld8(n0 +      hi * 16), ld8(n0 +      hi * 16 + 8));
        v16bf nb01 = combine(ld8(n0 + 32 + hi * 16), ld8(n0 + 32 + hi * 16 + 8));
        v16bf nb10 = combine(ld8(n1 +      hi * 16), ld8(n1 +      hi * 16 + 8));
        v16bf nb11 = combine(ld8(n1 + 32 + hi * 16), ld8(n1 + 32 + hi * 16 + 8));

        const int cb0 = bkt[bh * SEQ + j0 + n16];
        const int cb1 = bkt[bh * SEQ + j0 + 16 + n16];

        // ---- mask + streaming softmax (per-row stats via 16-lane shuffles) --
#pragma unroll
        for (int r = 0; r < 8; ++r) {
            float s0 = (rbk[r] == cb0) ? S0[r] * SCALE : 0.f;
            float s1 = (rbk[r] == cb1) ? S1[r] * SCALE : 0.f;
            float t = fmaxf(s0, s1);
            t = fmaxf(t, __shfl_xor(t, 1, 32));
            t = fmaxf(t, __shfl_xor(t, 2, 32));
            t = fmaxf(t, __shfl_xor(t, 4, 32));
            t = fmaxf(t, __shfl_xor(t, 8, 32));
            float mnew  = fmaxf(mrow[r], t);
            float alpha = __expf(mrow[r] - mnew);
            float p0 = __expf(s0 - mnew);
            float p1 = __expf(s1 - mnew);
            float ps = p0 + p1;
            ps += __shfl_xor(ps, 1, 32);
            ps += __shfl_xor(ps, 2, 32);
            ps += __shfl_xor(ps, 4, 32);
            ps += __shfl_xor(ps, 8, 32);
            lrow[r] = lrow[r] * alpha + ps;
            mrow[r] = mnew;
            O0[r] *= alpha; O1[r] *= alpha; O2[r] *= alpha; O3[r] *= alpha;
            // stage P (bf16) to LDS in row-major [16][32] for A-frag reshape
            int m = r + 8 * hi;
            Pt[wv][m * 32 +      n16] = (bf16_t)p0;
            Pt[wv][m * 32 + 16 + n16] = (bf16_t)p1;
        }

        // ---- re-load P as a 16x32 bf16 A-fragment (ds b128 loads) ----------
        v16bf pf = combine(
            *reinterpret_cast<const v8bf*>(&Pt[wv][n16 * 32 + hi * 16]),
            *reinterpret_cast<const v8bf*>(&Pt[wv][n16 * 32 + hi * 16 + 8]));

        // ---- O += P @ V ----------------------------------------------------
        O0 = wmma_bf16(pf, vb0, O0);
        O1 = wmma_bf16(pf, vb1, O1);
        O2 = wmma_bf16(pf, vb2, O2);
        O3 = wmma_bf16(pf, vb3, O3);

        b00 = nb00; b01 = nb01; b10 = nb10; b11 = nb11;
    }

    // ---- epilogue: normalize by row sums, write f32 output ------------------
#pragma unroll
    for (int r = 0; r < 8; ++r) {
        float inv = 1.f / lrow[r];
        int row = rbase + r + 8 * hi;
        float* op = out + ((b * SEQ + row) * DIM + h * EDIM);
        op[ 0 + n16] = O0[r] * inv;
        op[16 + n16] = O1[r] * inv;
        op[32 + n16] = O2[r] * inv;
        op[48 + n16] = O3[r] * inv;
    }
}

// ---------------------------------------------------------------------------
// Workspace layout (bytes):
//   xb   @ 0         : 8,388,608   (x bf16)
//   wqb  @ 8388608   :   524,288
//   wvb  @ 8912896   :   524,288
//   qf   @ 9437184   : 16,777,216  (q f32, for exact LSH sign bits)
//   qb   @ 26214400  : 8,388,608   (q bf16)
//   vT   @ 34603008  : 8,388,608   (v bf16 transposed [B,H,e,N])
//   bkt  @ 42991616  :   262,144   (int32 buckets [B,H,N])
// total ~41.3 MB
// ---------------------------------------------------------------------------
extern "C" void kernel_launch(void* const* d_in, const int* in_sizes, int n_in,
                              void* d_out, int out_size, void* d_ws, size_t ws_size,
                              hipStream_t stream) {
    const float* x   = (const float*)d_in[0];
    const float* Wq  = (const float*)d_in[1];
    const float* bq  = (const float*)d_in[2];
    const float* Wv  = (const float*)d_in[3];
    const float* bv  = (const float*)d_in[4];
    const float* hyp = (const float*)d_in[5];
    float* out = (float*)d_out;

    char* ws = (char*)d_ws;
    bf16_t* xb  = (bf16_t*)(ws);
    bf16_t* wqb = (bf16_t*)(ws + 8388608);
    bf16_t* wvb = (bf16_t*)(ws + 8912896);
    float*  qf  = (float*) (ws + 9437184);
    bf16_t* qb  = (bf16_t*)(ws + 26214400);
    bf16_t* vT  = (bf16_t*)(ws + 34603008);
    int*    bkt = (int*)   (ws + 42991616);

    const int nx = BATCH * SEQ * DIM;   // 4,194,304
    const int nw = DIM * DIM;           //   262,144

    cvt_bf16_kernel<<<(nx + 255) / 256, 256, 0, stream>>>(x,  xb,  nx);
    cvt_bf16_kernel<<<(nw + 255) / 256, 256, 0, stream>>>(Wq, wqb, nw);
    cvt_bf16_kernel<<<(nw + 255) / 256, 256, 0, stream>>>(Wv, wvb, nw);

    // 4096 wave-tiles (512 M-tiles x 8 D-tiles) / 8 waves per block
    gemm_bias_kernel<<<512, 256, 0, stream>>>(xb, wqb, bq, qf, qb, nullptr, 0);
    gemm_bias_kernel<<<512, 256, 0, stream>>>(xb, wvb, bv, nullptr, nullptr, vT, 1);

    buckets_kernel<<<(BATCH * SEQ * HEADS + 255) / 256, 256, 0, stream>>>(qf, hyp, bkt);

    // 32 (b,h) x 16 row-blocks of 128 rows (8 waves x 16 rows)
    lsh_attn_kernel<<<512, 256, 0, stream>>>(qb, vT, bkt, out);
}
